// Model_50620484551057
// MI455X (gfx1250) — matrix operand order repaired and checked by reading, not compiled, not run on hardware
//
#include <hip/hip_runtime.h>

#define NN   10000
#define DD   1024
#define HH   8
#define CC   128
#define EE   160000
#define ETOT (EE + NN)
#define NGR  64
#define KT   32      // DD/32
#define MT   625     // NN/16
#define NEG_SLOPE 0.2f
#define BN_EPS 1e-5f

typedef __attribute__((ext_vector_type(16))) __bf16 v16bf;
typedef __attribute__((ext_vector_type(8)))  float  v8f;

__device__ __forceinline__ unsigned short f2bf(float f) {
  unsigned u = __float_as_uint(f);
  u += 0x7FFFu + ((u >> 16) & 1u);          // round-to-nearest-even
  return (unsigned short)(u >> 16);
}

__device__ __forceinline__ void atomicMaxF(float* a, float v) {
  if (v >= 0.f) atomicMax((int*)a, __float_as_int(v));
  else          atomicMin((unsigned int*)a, __float_as_uint(v));
}

__device__ __forceinline__ void edge_sd(const long long* ei, int e, int& s, int& d) {
  if (e < EE) { s = (int)ei[e]; d = (int)ei[EE + e]; }
  else        { s = d = e - EE; }
}

// ---------- pack A: f32 [NN,DD] row-major -> bf16 WMMA A-fragments ----------
// tile t = mt*KT + kt, fragment = 32 lanes x 16 bf16 (32B/lane, contiguous)
// A 16x32 layout: lane = 16g+m holds row m; elem e=2v+w -> K = ((v<4)?0:16)+8g+2(v&3)+w
__global__ void k_pack_a(const float* __restrict__ src, unsigned short* __restrict__ dst) {
  int t    = blockIdx.x * 8 + (threadIdx.x >> 5);
  int lane = threadIdx.x & 31;
  int mt = t / KT, kt = t % KT;
  int g = lane >> 4, m = lane & 15;
  const float* rowp = src + (size_t)(mt * 16 + m) * DD + kt * 32;
  unsigned short vals[16];
#pragma unroll
  for (int e = 0; e < 16; ++e) {
    int v = e >> 1, w = e & 1;
    int k = ((v < 4) ? 0 : 16) + 8 * g + 2 * (v & 3) + w;
    vals[e] = f2bf(rowp[k]);
  }
  uint4* o = (uint4*)(dst + (size_t)t * 512 + lane * 16);
  o[0] = ((uint4*)vals)[0];
  o[1] = ((uint4*)vals)[1];
}

// ---------- pack B: W f32 [DD,DD] (k-major) -> bf16 WMMA B-fragments ----------
// tile t = nt*KT + kt ; B 32x16 layout: lane = 16g+n holds col n; elem e -> K = ((e<8)?0:16)+8g+(e&7)
__global__ void k_pack_b(const float* __restrict__ W, unsigned short* __restrict__ dst) {
  int t    = blockIdx.x * 8 + (threadIdx.x >> 5);
  int lane = threadIdx.x & 31;
  int nt = t / KT, kt = t % KT;
  int g = lane >> 4, n = lane & 15;
  int col = nt * 16 + n;
  unsigned short vals[16];
#pragma unroll
  for (int e = 0; e < 16; ++e) {
    int k = kt * 32 + ((e < 8) ? 0 : 16) + 8 * g + (e & 7);
    vals[e] = f2bf(W[(size_t)k * DD + col]);
  }
  uint4* o = (uint4*)(dst + (size_t)t * 512 + lane * 16);
  o[0] = ((uint4*)vals)[0];
  o[1] = ((uint4*)vals)[1];
}

// ---------- GEMM: hproj = h @ W, bf16 WMMA, f32 accumulate ----------
// 8 waves/block, each wave: one 16-row strip x 64 cols (4 x 16x16 tiles)
__global__ void k_gemm(const unsigned short* __restrict__ pA,
                       const unsigned short* __restrict__ pB,
                       float* __restrict__ out) {
  int gw   = blockIdx.x * 8 + (threadIdx.x >> 5);
  int lane = threadIdx.x & 31;
  int mt  = gw >> 4;
  int nt0 = (gw & 15) * 4;

  const v16bf* A = (const v16bf*)pA;
  const v16bf* B = (const v16bf*)pB;

  v8f acc[4] = {v8f{}, v8f{}, v8f{}, v8f{}};
  for (int kt = 0; kt < KT; ++kt) {
    v16bf a = A[(size_t)(mt * KT + kt) * 32 + lane];
#pragma unroll
    for (int j = 0; j < 4; ++j) {
      v16bf b = B[(size_t)((nt0 + j) * KT + kt) * 32 + lane];
      acc[j] = __builtin_amdgcn_wmma_f32_16x16x32_bf16(
          false, a, false, b, (short)0, acc[j], false, false);
    }
  }
  int g = lane >> 4, n = lane & 15;
#pragma unroll
  for (int j = 0; j < 4; ++j)
#pragma unroll
    for (int v = 0; v < 8; ++v)
      out[(size_t)(mt * 16 + v + 8 * g) * DD + (nt0 + j) * 16 + n] = acc[j][v];
}

// ---------- attention logits per node + init of m / denom / agg ----------
__global__ void k_alpha_init(const float* __restrict__ hproj,
                             const float* __restrict__ attS,
                             const float* __restrict__ attD,
                             float* __restrict__ aS, float* __restrict__ aD,
                             float* __restrict__ mM, float* __restrict__ dn,
                             float* __restrict__ agg) {
  int n = blockIdx.x, t = threadIdx.x;
  int h = t >> 5;
  int c0 = h * CC + (t & 31) * 4;
  const float* hp = hproj + (size_t)n * DD + c0;
  const float* as = attS + c0;
  const float* ad = attD + c0;
  float s1 = 0.f, s2 = 0.f;
#pragma unroll
  for (int k = 0; k < 4; ++k) { s1 += hp[k] * as[k]; s2 += hp[k] * ad[k]; }
  for (int off = 16; off; off >>= 1) {
    s1 += __shfl_down(s1, off);
    s2 += __shfl_down(s2, off);
  }
  if ((t & 31) == 0) { aS[n * HH + h] = s1; aD[n * HH + h] = s2; }
  if (t < HH) { mM[n * HH + t] = -1e30f; dn[n * HH + t] = 0.f; }
  float4 z = {0.f, 0.f, 0.f, 0.f};
  ((float4*)(agg + (size_t)n * DD))[t] = z;
}

__global__ void k_edge_max(const long long* __restrict__ ei,
                           const float* __restrict__ aS, const float* __restrict__ aD,
                           float* __restrict__ mM) {
  int idx = blockIdx.x * 256 + threadIdx.x;
  if (idx >= ETOT * HH) return;
  int e = idx >> 3, h = idx & 7;
  int s, d; edge_sd(ei, e, s, d);
  float v = aS[s * HH + h] + aD[d * HH + h];
  v = v > 0.f ? v : NEG_SLOPE * v;
  atomicMaxF(&mM[d * HH + h], v);
}

__global__ void k_edge_den(const long long* __restrict__ ei,
                           const float* __restrict__ aS, const float* __restrict__ aD,
                           const float* __restrict__ mM, float* __restrict__ dn) {
  int idx = blockIdx.x * 256 + threadIdx.x;
  if (idx >= ETOT * HH) return;
  int e = idx >> 3, h = idx & 7;
  int s, d; edge_sd(ei, e, s, d);
  float v = aS[s * HH + h] + aD[d * HH + h];
  v = v > 0.f ? v : NEG_SLOPE * v;
  atomicAdd(&dn[d * HH + h], __expf(v - mM[d * HH + h]));
}

// one 256-thread block per edge: scatter alpha * h[src] into agg[dst]
__global__ void k_edge_agg(const long long* __restrict__ ei,
                           const float* __restrict__ aS, const float* __restrict__ aD,
                           const float* __restrict__ mM, const float* __restrict__ dn,
                           const float* __restrict__ hproj, float* __restrict__ agg) {
  __shared__ float al[HH];
  __shared__ int ssrc, sdst;
  int e = blockIdx.x, t = threadIdx.x;
  if (t < HH) {
    int s, d; edge_sd(ei, e, s, d);
    float v = aS[s * HH + t] + aD[d * HH + t];
    v = v > 0.f ? v : NEG_SLOPE * v;
    al[t] = __expf(v - mM[d * HH + t]) / dn[d * HH + t];
    if (t == 0) { ssrc = s; sdst = d; }
  }
  __syncthreads();
  int c = t * 4;
  float a = al[t >> 5];
  const float* hs = hproj + (size_t)ssrc * DD + c;
  float* od = agg + (size_t)sdst * DD + c;
  atomicAdd(&od[0], a * hs[0]);
  atomicAdd(&od[1], a * hs[1]);
  atomicAdd(&od[2], a * hs[2]);
  atomicAdd(&od[3], a * hs[3]);
}

__global__ void k_zero(float* __restrict__ p, int n) {
  int i = blockIdx.x * 256 + threadIdx.x;
  if (i < n) p[i] = 0.f;
}

// partial per-column sums of pre = prev + agg + bias  (coalesced, reg accum)
__global__ void k_bn_partial(const float* __restrict__ prev, const float* __restrict__ agg,
                             const float* __restrict__ bias,
                             float* __restrict__ sum, float* __restrict__ sumsq) {
  int c0 = threadIdx.x * 4;
  float b0 = bias[c0], b1 = bias[c0 + 1], b2 = bias[c0 + 2], b3 = bias[c0 + 3];
  float s0 = 0, s1 = 0, s2 = 0, s3 = 0, q0 = 0, q1 = 0, q2 = 0, q3 = 0;
  for (int r = blockIdx.x; r < NN; r += gridDim.x) {
    size_t base = (size_t)r * DD + c0;
    float p0 = prev[base] + agg[base] + b0;
    float p1 = prev[base + 1] + agg[base + 1] + b1;
    float p2 = prev[base + 2] + agg[base + 2] + b2;
    float p3 = prev[base + 3] + agg[base + 3] + b3;
    s0 += p0; s1 += p1; s2 += p2; s3 += p3;
    q0 += p0 * p0; q1 += p1 * p1; q2 += p2 * p2; q3 += p3 * p3;
  }
  atomicAdd(&sum[c0], s0);   atomicAdd(&sum[c0 + 1], s1);
  atomicAdd(&sum[c0 + 2], s2); atomicAdd(&sum[c0 + 3], s3);
  atomicAdd(&sumsq[c0], q0);   atomicAdd(&sumsq[c0 + 1], q1);
  atomicAdd(&sumsq[c0 + 2], q2); atomicAdd(&sumsq[c0 + 3], q3);
}

__global__ void k_bn_final(const float* __restrict__ sum, const float* __restrict__ sumsq,
                           float* __restrict__ mu, float* __restrict__ rs) {
  int d = blockIdx.x * 256 + threadIdx.x;
  if (d >= DD) return;
  float m = sum[d] / (float)NN;
  float v = sumsq[d] / (float)NN - m * m;
  mu[d] = m;
  rs[d] = rsqrtf(v + BN_EPS);
}

__global__ void k_bn_apply(const float* __restrict__ prev, const float* __restrict__ agg,
                           const float* __restrict__ bias,
                           const float* __restrict__ mu, const float* __restrict__ rs,
                           const float* __restrict__ gamma, const float* __restrict__ beta,
                           float* __restrict__ out) {
  size_t base = ((size_t)blockIdx.x * 256 + threadIdx.x) * 4;
  int col = (int)(base & (DD - 1));
#pragma unroll
  for (int k = 0; k < 4; ++k) {
    float pre = prev[base + k] + agg[base + k] + bias[col + k];
    float v = (pre - mu[col + k]) * rs[col + k] * gamma[col + k] + beta[col + k];
    out[base + k] = fmaxf(v, 0.f);
  }
}

__global__ void k_pool(const float* __restrict__ h, const long long* __restrict__ batch,
                       float* __restrict__ psum, float* __restrict__ pcnt) {
  int n = blockIdx.x, t = threadIdx.x;
  int b = (int)batch[n];
  int c = t * 4;
  const float* hp = h + (size_t)n * DD + c;
  float* pp = psum + (size_t)b * DD + c;
  atomicAdd(&pp[0], hp[0]); atomicAdd(&pp[1], hp[1]);
  atomicAdd(&pp[2], hp[2]); atomicAdd(&pp[3], hp[3]);
  if (t == 0) atomicAdd(&pcnt[b], 1.f);
}

__global__ void k_final(const float* __restrict__ psum, const float* __restrict__ pcnt,
                        const float* __restrict__ Wout, const float* __restrict__ bout,
                        float* __restrict__ out) {
  __shared__ float r0[256], r1[256];
  int g = blockIdx.x, t = threadIdx.x;
  float p0 = 0.f, p1 = 0.f;
  for (int d = t; d < DD; d += 256) {
    float p = psum[(size_t)g * DD + d];
    p0 += p * Wout[d * 2];
    p1 += p * Wout[d * 2 + 1];
  }
  r0[t] = p0; r1[t] = p1;
  __syncthreads();
  for (int s = 128; s; s >>= 1) {
    if (t < s) { r0[t] += r0[t + s]; r1[t] += r1[t + s]; }
    __syncthreads();
  }
  if (t == 0) {
    float cnt = fmaxf(pcnt[g], 1.f);
    out[g * 2]     = r0[0] / cnt + bout[0];
    out[g * 2 + 1] = r1[0] / cnt + bout[1];
  }
}

extern "C" void kernel_launch(void* const* d_in, const int* in_sizes, int n_in,
                              void* d_out, int out_size, void* d_ws, size_t ws_size,
                              hipStream_t stream) {
  (void)in_sizes; (void)n_in; (void)out_size; (void)ws_size;
  const float*     x     = (const float*)d_in[0];
  const long long* ei    = (const long long*)d_in[1];
  const long long* batch = (const long long*)d_in[2];
  const float*     W     = (const float*)d_in[3];   // [3,1024,1024]
  const float*     attS  = (const float*)d_in[4];   // [3,8,128]
  const float*     attD  = (const float*)d_in[5];
  const float*     attB  = (const float*)d_in[6];   // [3,1024]
  const float*     gamma = (const float*)d_in[7];
  const float*     beta  = (const float*)d_in[8];
  const float*     Wout  = (const float*)d_in[9];   // [1024,2]
  const float*     bout  = (const float*)d_in[10];  // [2]
  float* out = (float*)d_out;

  char* ws = (char*)d_ws;
  size_t off = 0;
  auto take = [&](size_t bytes) -> char* {
    char* p = ws + off;
    off += (bytes + 255) & ~(size_t)255;
    return p;
  };
  float* buf0  = (float*)take(sizeof(float) * (size_t)NN * DD);
  float* buf1  = (float*)take(sizeof(float) * (size_t)NN * DD);
  float* buf2  = (float*)take(sizeof(float) * (size_t)NN * DD);
  float* hproj = (float*)take(sizeof(float) * (size_t)NN * DD);
  unsigned short* pA = (unsigned short*)take(2ull * NN * DD);
  unsigned short* pB = (unsigned short*)take(2ull * DD * DD);
  float* aS  = (float*)take(4ull * NN * HH);
  float* aD  = (float*)take(4ull * NN * HH);
  float* mM  = (float*)take(4ull * NN * HH);
  float* dn  = (float*)take(4ull * NN * HH);
  float* bnS = (float*)take(4ull * 2 * DD);     // sum | sumsq
  float* bnMu = (float*)take(4ull * DD);
  float* bnRs = (float*)take(4ull * DD);
  float* psum = (float*)take(4ull * (NGR * DD + NGR));
  float* pcnt = psum + (size_t)NGR * DD;

  const float* cur[3]  = { x,    buf0, buf1 };
  const float* prev[3] = { x,    x,    buf0 };
  float*       agg[3]  = { buf0, buf1, buf2 };

  int edgeBlocks = (ETOT * HH + 255) / 256;

  for (int i = 0; i < 3; ++i) {
    k_pack_a<<<(MT * KT) / 8, 256, 0, stream>>>(cur[i], pA);
    k_pack_b<<<((DD / 16) * KT) / 8, 256, 0, stream>>>(W + (size_t)i * DD * DD, pB);
    k_gemm<<<(MT * 16) / 8, 256, 0, stream>>>(pA, pB, hproj);

    k_alpha_init<<<NN, 256, 0, stream>>>(hproj, attS + i * HH * CC, attD + i * HH * CC,
                                         aS, aD, mM, dn, agg[i]);
    k_edge_max<<<edgeBlocks, 256, 0, stream>>>(ei, aS, aD, mM);
    k_edge_den<<<edgeBlocks, 256, 0, stream>>>(ei, aS, aD, mM, dn);
    k_edge_agg<<<ETOT, 256, 0, stream>>>(ei, aS, aD, mM, dn, hproj, agg[i]);

    k_zero<<<(2 * DD + 255) / 256, 256, 0, stream>>>(bnS, 2 * DD);
    k_bn_partial<<<256, 256, 0, stream>>>(prev[i], agg[i], attB + (size_t)i * DD,
                                          bnS, bnS + DD);
    k_bn_final<<<(DD + 255) / 256, 256, 0, stream>>>(bnS, bnS + DD, bnMu, bnRs);
    k_bn_apply<<<(NN * DD) / 1024, 256, 0, stream>>>(prev[i], agg[i], attB + (size_t)i * DD,
                                                     bnMu, bnRs,
                                                     gamma + (size_t)i * DD,
                                                     beta + (size_t)i * DD, agg[i]);
  }

  k_zero<<<(NGR * DD + NGR + 255) / 256, 256, 0, stream>>>(psum, NGR * DD + NGR);
  k_pool<<<NN, 256, 0, stream>>>(buf2, batch, psum, pcnt);
  k_final<<<NGR, 256, 0, stream>>>(psum, pcnt, Wout, bout, out);
}
